// GPT_59141699666120
// MI455X (gfx1250) — compile-verified, matrix operand order checked by reading, not run on hardware
//
#include <hip/hip_runtime.h>
#include <hip/hip_bf16.h>
#include <math.h>

// ---------------------------------------------------------------------------
// GPT-2 small forward on MI455X (gfx1250): all GEMMs via v_wmma_f32_16x16x32_f16
// Block tile 128x128, 8 waves (4Mx2N), wave tile 32x64 -> 8 WMMA / 6 frag loads.
// ---------------------------------------------------------------------------

typedef __attribute__((ext_vector_type(16))) _Float16 v16h;
typedef __attribute__((ext_vector_type(8)))  _Float16 v8h;
typedef __attribute__((ext_vector_type(8)))  float    v8f;

#define GPT_L   6
#define GPT_C   768
#define GPT_H   12
#define GPT_HD  64
#define GPT_T   1024
#define GPT_B   2
#define GPT_M   (GPT_B * GPT_T)   // 2048 rows
#define GPT_V   50257
#define ATT_SCALE 0.125f          // 1/sqrt(64)

// ------------------------------ helpers ------------------------------------

__device__ __forceinline__ v16h load_frag(const _Float16* p) {
  // A/B fragment for 16x16x32 f16 WMMA: per lane, 8 halfs at K-base and 8 at K-base+16
  union { v16h v; v8h h[2]; } f;
  f.h[0] = *(const v8h*)(p);
  f.h[1] = *(const v8h*)(p + 16);
  return f.v;
}

__device__ __forceinline__ float half_max16(float v) {
  // reduce across the 16-lane half (lanes 0-15 or 16-31) of a wave32
  v = fmaxf(v, __shfl_xor(v, 1, 32));
  v = fmaxf(v, __shfl_xor(v, 2, 32));
  v = fmaxf(v, __shfl_xor(v, 4, 32));
  v = fmaxf(v, __shfl_xor(v, 8, 32));
  return v;
}
__device__ __forceinline__ float half_sum16(float v) {
  v += __shfl_xor(v, 1, 32);
  v += __shfl_xor(v, 2, 32);
  v += __shfl_xor(v, 4, 32);
  v += __shfl_xor(v, 8, 32);
  return v;
}

// ------------------------- elementwise kernels ------------------------------

__global__ __launch_bounds__(256) void cvt_f16_kernel(const float* __restrict__ in,
                                                      _Float16* __restrict__ out,
                                                      size_t n) {
  size_t i = (size_t)blockIdx.x * blockDim.x + threadIdx.x;
  size_t stride = (size_t)gridDim.x * blockDim.x;
  for (; i < n; i += stride) out[i] = (_Float16)in[i];
}

__global__ __launch_bounds__(256) void embed_kernel(const int* __restrict__ idx,
                                                    const float* __restrict__ tok,
                                                    const float* __restrict__ pos,
                                                    float* __restrict__ x) {
  const int m = blockIdx.x;                 // 0..2047  (b*1024 + t)
  const int t = m & (GPT_T - 1);
  const int token = idx[m];
  const float* tr = tok + (size_t)token * GPT_C;
  const float* pr = pos + (size_t)t * GPT_C;
  float* xr = x + (size_t)m * GPT_C;
#pragma unroll
  for (int i = 0; i < 3; ++i) {
    int c = threadIdx.x + i * 256;
    xr[c] = tr[c] + pr[c];
  }
}

__global__ __launch_bounds__(256) void layernorm_f16_kernel(const float* __restrict__ x,
                                                            const float* __restrict__ w,
                                                            const float* __restrict__ bb,
                                                            _Float16* __restrict__ out) {
  const int rowi = blockIdx.x;
  const int tid = threadIdx.x;
  const float* xr = x + (size_t)rowi * GPT_C;
  float s = 0.f, ss = 0.f;
#pragma unroll
  for (int i = 0; i < 3; ++i) {
    float v = xr[tid + i * 256];
    s += v;
    ss += v * v;
  }
  __shared__ float rs[256], rq[256];
  rs[tid] = s; rq[tid] = ss;
  __syncthreads();
  for (int st = 128; st > 0; st >>= 1) {
    if (tid < st) { rs[tid] += rs[tid + st]; rq[tid] += rq[tid + st]; }
    __syncthreads();
  }
  const float mean = rs[0] * (1.0f / GPT_C);
  const float var  = rq[0] * (1.0f / GPT_C) - mean * mean;
  const float inv  = rsqrtf(var + 1e-5f);
  _Float16* orow = out + (size_t)rowi * GPT_C;
#pragma unroll
  for (int i = 0; i < 3; ++i) {
    int c = tid + i * 256;
    orow[c] = (_Float16)((xr[c] - mean) * inv * w[c] + bb[c]);
  }
}

// ------------------------------ WMMA GEMM -----------------------------------
// C[M,N] = A[M,K] @ W[N,K]^T + bias, f16 inputs, f32 accumulate.
// Block tile 128(M) x 128(N), K-step 32, 8 waves in 4(M)x2(N), wave tile 32x64.

enum { EPI_QKV = 0, EPI_RES = 1, EPI_GELU = 2, EPI_OUT = 3 };

__device__ __forceinline__ void stage_tiles(const _Float16* __restrict__ A,
                                            const _Float16* __restrict__ W,
                                            int K, int N, int m0, int n0, int k0,
                                            int tid,
                                            _Float16 (*As)[40], _Float16 (*Bs)[40]) {
  // A tile 128x32 halfs: 512 chunks of 8 halfs, 2 per thread
#pragma unroll
  for (int i = 0; i < 2; ++i) {
    int c = tid + i * 256;
    int r = c >> 2, cc = (c & 3) << 3;
    *(v8h*)&As[r][cc] = *(const v8h*)&A[(size_t)(m0 + r) * K + k0 + cc];
  }
  // B tile 128x32 halfs: 512 chunks, 2 per thread (N-bounds guarded for lm_head)
#pragma unroll
  for (int i = 0; i < 2; ++i) {
    int c = tid + i * 256;
    int r = c >> 2, cc = (c & 3) << 3;
    if (n0 + r < N) {
      const _Float16* src = &W[(size_t)(n0 + r) * K + k0 + cc];
      *(v8h*)&Bs[r][cc] = *(const v8h*)src;
      __builtin_prefetch(src + 64, 0, 1);   // global_prefetch next-next K tile
    } else {
      v8h z = {};
      *(v8h*)&Bs[r][cc] = z;
    }
  }
}

template <int EPI>
__global__ __launch_bounds__(256, 1)
void gemm_wmma_kernel(const _Float16* __restrict__ A, const _Float16* __restrict__ W,
                      const float* __restrict__ bias, int N, int K,
                      float* __restrict__ outf, const float* __restrict__ res,
                      _Float16* __restrict__ outh,
                      _Float16* __restrict__ q16, _Float16* __restrict__ k16,
                      _Float16* __restrict__ v16) {
  __shared__ _Float16 As[2][128][40];   // padded stride: 80B rows, 16B aligned
  __shared__ _Float16 Bs[2][128][40];

  const int tid = threadIdx.x;
  const int wave = tid >> 5, lane = tid & 31;
  const int wm = (wave & 3) * 32, wn = (wave >> 2) * 64;
  const int m0 = blockIdx.y * 128, n0 = blockIdx.x * 128;
  const int row = lane & 15, ks = (lane >> 4) << 3;

  v8f acc[2][4] = {};

  const int nk = K >> 5;
  stage_tiles(A, W, K, N, m0, n0, 0, tid, As[0], Bs[0]);
  __syncthreads();
  for (int kt = 0; kt < nk; ++kt) {
    const int cur = kt & 1;
    if (kt + 1 < nk)
      stage_tiles(A, W, K, N, m0, n0, (kt + 1) << 5, tid, As[cur ^ 1], Bs[cur ^ 1]);
    v16h a0 = load_frag(&As[cur][wm + row][ks]);
    v16h a1 = load_frag(&As[cur][wm + 16 + row][ks]);
    v16h b[4];
#pragma unroll
    for (int j = 0; j < 4; ++j)
      b[j] = load_frag(&Bs[cur][wn + j * 16 + row][ks]);
#pragma unroll
    for (int j = 0; j < 4; ++j) {
      acc[0][j] = __builtin_amdgcn_wmma_f32_16x16x32_f16(false, a0, false, b[j], (short)0, acc[0][j], false, false);
      acc[1][j] = __builtin_amdgcn_wmma_f32_16x16x32_f16(false, a1, false, b[j], (short)0, acc[1][j], false, false);
    }
    __syncthreads();
  }

  // Epilogue. D layout: vgpr r -> M = r + 8*(lane>=16); N = lane&15.
#pragma unroll
  for (int ti = 0; ti < 2; ++ti) {
#pragma unroll
    for (int tj = 0; tj < 4; ++tj) {
      const int n = n0 + wn + tj * 16 + (lane & 15);
      if (n >= N) continue;
      const int mb = m0 + wm + ti * 16 + ((lane >> 4) << 3);
      const float bv = bias ? bias[n] : 0.0f;
#pragma unroll
      for (int r = 0; r < 8; ++r) {
        const int m = mb + r;
        float val = acc[ti][tj][r] + bv;
        if (EPI == EPI_QKV) {
          const int which = n / GPT_C, rem = n % GPT_C;
          const int hh = rem >> 6, dd = rem & 63;
          const int b = m >> 10, t = m & (GPT_T - 1);
          const size_t di = (((size_t)(b * GPT_H + hh) * GPT_T) + t) * GPT_HD + dd;
          if (which == 0)      q16[di] = (_Float16)(val * ATT_SCALE);
          else if (which == 1) k16[di] = (_Float16)val;
          else                 v16[di] = (_Float16)val;
        } else if (EPI == EPI_RES) {
          outf[(size_t)m * N + n] = res[(size_t)m * N + n] + val;
        } else if (EPI == EPI_GELU) {
          outh[(size_t)m * N + n] =
              (_Float16)(0.5f * val * (1.0f + erff(val * 0.70710678118f)));
        } else {
          outf[(size_t)m * N + n] = val;
        }
      }
    }
  }
}

// --------------------------- flash attention --------------------------------
// grid (B*H=24, T/64=16), 128 threads = 4 waves; wave w owns q rows qt*64+w*16..+15.
// Streams keys in blocks of 32; S = Q.K^T (2 tiles x 2 WMMAs), online softmax,
// P re-laid out through per-wave LDS, O += P.V (4 WMMAs over 64 head dims).

__global__ __launch_bounds__(128, 1)
void attn_wmma_kernel(const _Float16* __restrict__ q, const _Float16* __restrict__ k,
                      const _Float16* __restrict__ v, _Float16* __restrict__ out) {
  const int bh = blockIdx.x;
  const int qt = blockIdx.y;
  const int wave = threadIdx.x >> 5;
  const int lane = threadIdx.x & 31;
  const int qbase = qt * 64 + wave * 16;

  const size_t base = (size_t)bh * GPT_T * GPT_HD;
  const _Float16* Qp = q + base;
  const _Float16* Kp = k + base;
  const _Float16* Vp = v + base;

  __shared__ _Float16 Ps[4][16][40];     // per-wave P tile  (D-layout -> A-layout relay)
  __shared__ _Float16 Vt[4][64][40];     // per-wave V^T tile [d][key]

  const int row = lane & 15, ks = (lane >> 4) << 3;
  const int mrow = ((lane >> 4) << 3);   // +0 or +8 row offset in D layout

  // Q fragments (rows qbase..qbase+15, d-chunks 0 and 32), pre-scaled by 1/8
  v16h qf[2];
#pragma unroll
  for (int c = 0; c < 2; ++c)
    qf[c] = load_frag(&Qp[(size_t)(qbase + row) * GPT_HD + c * 32 + ks]);

  float mi[8], li[8];
#pragma unroll
  for (int r = 0; r < 8; ++r) { mi[r] = -1e30f; li[r] = 0.0f; }
  v8f accO[4] = {};

  const int nkb = (qbase + 16 + 31) >> 5;   // 32-key blocks needed (causal)
  for (int kb = 0; kb < nkb; ++kb) {
    const int kk = kb << 5;

    // ---- S = Q.K^T for two 16-key tiles
    v8f s[2] = {};
#pragma unroll
    for (int j = 0; j < 2; ++j) {
      const int kt0 = kk + j * 16;
#pragma unroll
      for (int c = 0; c < 2; ++c) {
        v16h bf = load_frag(&Kp[(size_t)(kt0 + row) * GPT_HD + c * 32 + ks]);
        s[j] = __builtin_amdgcn_wmma_f32_16x16x32_f16(false, qf[c], false, bf, (short)0, s[j], false, false);
      }
    }

    // ---- causal mask (uniform branch per wave)
    if (kk + 31 > qbase) {
#pragma unroll
      for (int j = 0; j < 2; ++j) {
        const int key = kk + j * 16 + (lane & 15);
#pragma unroll
        for (int r = 0; r < 8; ++r) {
          const int qr = qbase + r + mrow;
          if (key > qr) s[j][r] = -1e30f;
        }
      }
    }

    // ---- online softmax stats (rows live across lanes 0-15 / 16-31 halves)
    float al[8], sum[8];
#pragma unroll
    for (int r = 0; r < 8; ++r) {
      float mx = half_max16(fmaxf(s[0][r], s[1][r]));
      float mn = fmaxf(mi[r], mx);
      al[r] = __expf(mi[r] - mn);
      mi[r] = mn;
      float p0 = __expf(s[0][r] - mn);
      float p1 = __expf(s[1][r] - mn);
      sum[r] = p0 + p1;
      Ps[wave][r + mrow][(lane & 15)]      = (_Float16)p0;
      Ps[wave][r + mrow][16 + (lane & 15)] = (_Float16)p1;
    }
#pragma unroll
    for (int r = 0; r < 8; ++r) {
      li[r] = li[r] * al[r] + half_sum16(sum[r]);
#pragma unroll
      for (int dt = 0; dt < 4; ++dt) accO[dt][r] *= al[r];
    }

    // ---- stage V block transposed into per-wave LDS: Vt[d][key]
    for (int i = 0; i < 32; ++i) {
      const unsigned int pv = *(const unsigned int*)&Vp[(size_t)(kk + i) * GPT_HD + lane * 2];
      Vt[wave][lane * 2][i]     = ((const _Float16*)&pv)[0];
      Vt[wave][lane * 2 + 1][i] = ((const _Float16*)&pv)[1];
    }

    // ---- O += P.V : A-frag of P from LDS, B-frags of V^T from LDS
    v16h pf = load_frag(&Ps[wave][row][ks]);
#pragma unroll
    for (int dt = 0; dt < 4; ++dt) {
      v16h vf = load_frag(&Vt[wave][dt * 16 + row][ks]);
      accO[dt] = __builtin_amdgcn_wmma_f32_16x16x32_f16(false, pf, false, vf, (short)0, accO[dt], false, false);
    }
  }

  // ---- normalize and scatter back to [m, c] = [b*1024+t, h*64+d] f16
  const int b = bh / GPT_H, hh = bh % GPT_H;
#pragma unroll
  for (int dt = 0; dt < 4; ++dt) {
#pragma unroll
    for (int r = 0; r < 8; ++r) {
      const float val = accO[dt][r] / li[r];
      const int t = qbase + r + mrow;
      const int c = hh * GPT_HD + dt * 16 + (lane & 15);
      out[(size_t)(b * GPT_T + t) * GPT_C + c] = (_Float16)val;
    }
  }
}

// ------------------------------- driver -------------------------------------

extern "C" void kernel_launch(void* const* d_in, const int* in_sizes, int n_in,
                              void* d_out, int out_size, void* d_ws, size_t ws_size,
                              hipStream_t stream) {
  (void)in_sizes; (void)n_in; (void)out_size; (void)ws_size;

  const int*   idx      = (const int*)d_in[0];
  const float* tok_emb  = (const float*)d_in[1];
  const float* pos_emb  = (const float*)d_in[2];
  const float* ln1_w    = (const float*)d_in[3];
  const float* ln1_b    = (const float*)d_in[4];
  const float* qkv_w    = (const float*)d_in[5];
  const float* qkv_b    = (const float*)d_in[6];
  const float* proj_w   = (const float*)d_in[7];
  const float* proj_b   = (const float*)d_in[8];
  const float* ln2_w    = (const float*)d_in[9];
  const float* ln2_b    = (const float*)d_in[10];
  const float* fc1_w    = (const float*)d_in[11];
  const float* fc1_b    = (const float*)d_in[12];
  const float* fc2_w    = (const float*)d_in[13];
  const float* fc2_b    = (const float*)d_in[14];
  const float* lnf_w    = (const float*)d_in[15];
  const float* lnf_b    = (const float*)d_in[16];
  const float* lm_w     = (const float*)d_in[17];

  // ---- carve workspace (256B aligned regions)
  char* ws = (char*)d_ws;
  size_t off = 0;
  auto carve = [&](size_t bytes) -> void* {
    void* p = ws + off;
    off += (bytes + 255) & ~(size_t)255;
    return p;
  };
  const size_t QKVW = (size_t)GPT_L * 3 * GPT_C * GPT_C;
  const size_t PRJW = (size_t)GPT_L * GPT_C * GPT_C;
  const size_t FCW  = (size_t)GPT_L * 4 * GPT_C * GPT_C;
  const size_t LMW  = (size_t)GPT_V * GPT_C;
  const size_t ACT  = (size_t)GPT_M * GPT_C;       // 2048*768
  const size_t G4   = (size_t)GPT_M * 4 * GPT_C;

  _Float16* qkvw16 = (_Float16*)carve(QKVW * 2);
  _Float16* projw16 = (_Float16*)carve(PRJW * 2);
  _Float16* fc1w16 = (_Float16*)carve(FCW * 2);
  _Float16* fc2w16 = (_Float16*)carve(FCW * 2);
  _Float16* lm16   = (_Float16*)carve(LMW * 2);
  float*    x      = (float*)carve(ACT * 4);
  _Float16* h16    = (_Float16*)carve(ACT * 2);
  _Float16* q16    = (_Float16*)carve(ACT * 2);
  _Float16* k16    = (_Float16*)carve(ACT * 2);
  _Float16* v16    = (_Float16*)carve(ACT * 2);
  _Float16* att16  = (_Float16*)carve(ACT * 2);
  _Float16* g16    = (_Float16*)carve(G4 * 2);

  // ---- weight conversion fp32 -> f16
  cvt_f16_kernel<<<4096, 256, 0, stream>>>(qkv_w,  qkvw16, QKVW);
  cvt_f16_kernel<<<4096, 256, 0, stream>>>(proj_w, projw16, PRJW);
  cvt_f16_kernel<<<4096, 256, 0, stream>>>(fc1_w,  fc1w16, FCW);
  cvt_f16_kernel<<<4096, 256, 0, stream>>>(fc2_w,  fc2w16, FCW);
  cvt_f16_kernel<<<4096, 256, 0, stream>>>(lm_w,   lm16,   LMW);

  // ---- embeddings
  embed_kernel<<<GPT_M, 256, 0, stream>>>(idx, tok_emb, pos_emb, x);

  const dim3 gQKV(3 * GPT_C / 128, GPT_M / 128);    // (18,16)
  const dim3 gPRJ(GPT_C / 128, GPT_M / 128);        // (6,16)
  const dim3 gFC1(4 * GPT_C / 128, GPT_M / 128);    // (24,16)
  const dim3 gLM((GPT_V + 127) / 128, GPT_M / 128); // (393,16)
  const dim3 gATT(GPT_B * GPT_H, GPT_T / 64);       // (24,16)

  for (int l = 0; l < GPT_L; ++l) {
    layernorm_f16_kernel<<<GPT_M, 256, 0, stream>>>(x, ln1_w + l * GPT_C, ln1_b + l * GPT_C, h16);
    gemm_wmma_kernel<EPI_QKV><<<gQKV, 256, 0, stream>>>(
        h16, qkvw16 + (size_t)l * 3 * GPT_C * GPT_C, qkv_b + (size_t)l * 3 * GPT_C,
        3 * GPT_C, GPT_C, nullptr, nullptr, nullptr, q16, k16, v16);
    attn_wmma_kernel<<<gATT, 128, 0, stream>>>(q16, k16, v16, att16);
    gemm_wmma_kernel<EPI_RES><<<gPRJ, 256, 0, stream>>>(
        att16, projw16 + (size_t)l * GPT_C * GPT_C, proj_b + (size_t)l * GPT_C,
        GPT_C, GPT_C, x, x, nullptr, nullptr, nullptr, nullptr);
    layernorm_f16_kernel<<<GPT_M, 256, 0, stream>>>(x, ln2_w + l * GPT_C, ln2_b + l * GPT_C, h16);
    gemm_wmma_kernel<EPI_GELU><<<gFC1, 256, 0, stream>>>(
        h16, fc1w16 + (size_t)l * 4 * GPT_C * GPT_C, fc1_b + (size_t)l * 4 * GPT_C,
        4 * GPT_C, GPT_C, nullptr, nullptr, g16, nullptr, nullptr, nullptr);
    gemm_wmma_kernel<EPI_RES><<<gPRJ, 256, 0, stream>>>(
        g16, fc2w16 + (size_t)l * GPT_C * 4 * GPT_C, fc2_b + (size_t)l * GPT_C,
        GPT_C, 4 * GPT_C, x, x, nullptr, nullptr, nullptr, nullptr);
  }

  layernorm_f16_kernel<<<GPT_M, 256, 0, stream>>>(x, lnf_w, lnf_b, h16);
  gemm_wmma_kernel<EPI_OUT><<<gLM, 256, 0, stream>>>(
      h16, lm16, nullptr, GPT_V, GPT_C, (float*)d_out, nullptr, nullptr,
      nullptr, nullptr, nullptr);
}
